// Attention_35820027248841
// MI455X (gfx1250) — matrix operand order repaired and checked by reading, not compiled, not run on hardware
//
#include <hip/hip_runtime.h>
#include <cstdint>

// ---------------------------------------------------------------------------
// Flash attention for (b,h,d,n) layout on gfx1250 (MI455X), wave32 + WMMA.
// B=4, H=16, D=64, N=2048, fp32 in/out. f16 WMMA with f32 accumulation.
// Double-buffered GLOBAL_LOAD_ASYNC_TO_LDS staging (ASYNCcnt), exp2-domain
// online softmax, LDS-transposed epilogue for coalesced b128 stores.
// ---------------------------------------------------------------------------

typedef __attribute__((ext_vector_type(16))) _Float16 v16h;
typedef __attribute__((ext_vector_type(8)))  float    v8f;

union AFrag {                 // 16x32 f16 A-matrix fragment (8 VGPRs)
    v16h     v;
    _Float16 h[16];
    uint32_t u[8];
};
union CFrag {                 // 16x16 f32 C/D fragment (8 VGPRs)
    v8f   v;
    float f[8];
};

constexpr int kB  = 4;
constexpr int kH  = 16;
constexpr int kD  = 64;       // head dim
constexpr int kN  = 2048;     // sequence length
constexpr int kKT = 64;       // keys per LDS tile
constexpr int kIT = 128;      // queries per block (8 waves x 16 rows)

// LDS strides (padded vs. bank conflicts, keep 4B alignment for b32 loads)
constexpr int KT_STRIDE = 68; // halfs per j-row of K^T tile   [j][d]
constexpr int V_STRIDE  = 68; // halfs per d-row of V tile     [d][j]
constexpr int P_STRIDE  = 68; // halfs per i-row of P tile     [i][j]
constexpr int O_STRIDE  = 16; // floats per d-row of O tile    [d][i] (16B aligned)

// Shared-memory arena layout (bytes)
constexpr int TILE_F = kD * kKT;                                // floats per tile
constexpr int STAGE  = TILE_F * (int)sizeof(float);             // 16 KB fp32 tile
constexpr int OFF_K0 = 0;                                       // K buffers 0,1
constexpr int OFF_V0 = 2 * STAGE;                               // V buffers 0,1
constexpr int OFF_KT = 4 * STAGE;                               // f16 K^T tile
constexpr int OFF_VH = OFF_KT + kKT * KT_STRIDE * 2;            // f16 V tile
constexpr int OFF_P  = OFF_VH + kD * V_STRIDE * 2;              // per-wave P
constexpr int SMEM   = OFF_P + 8 * 16 * P_STRIDE * 2;           // ~98 KB total
// Epilogue O buffer (8 waves x 64 x 16 fp32 = 32 KB) reuses the stage region.

__global__ __launch_bounds__(256)
void fa_fwd_kernel(const float* __restrict__ qp,
                   const float* __restrict__ kp,
                   const float* __restrict__ vp,
                   float* __restrict__ outp)
{
    __shared__ __align__(16) unsigned char smem[SMEM];
    // Double buffers are contiguous: buffer b = base + b*TILE_F floats.
    float*    stageK = (float*)(smem + OFF_K0);
    float*    stageV = (float*)(smem + OFF_V0);
    _Float16* ldsKT  = (_Float16*)(smem + OFF_KT);
    _Float16* ldsVh  = (_Float16*)(smem + OFF_VH);
    _Float16* ldsP   = (_Float16*)(smem + OFF_P);
    float*    ldsO   = (float*)smem;                // reused after main loop

    const int t    = threadIdx.x;
    const int wave = t >> 5;
    const int lane = t & 31;
    const int lh   = lane >> 4;          // lane half (0: lanes 0-15, 1: 16-31)
    const int ln   = lane & 15;

    const int bh = blockIdx.x % (kB * kH);
    const int it = blockIdx.x / (kB * kH);
    const int i0 = it * kIT + wave * 16; // first query row owned by this wave

    const size_t base = (size_t)bh * kD * kN;       // per-head D x N matrix
    // Fold 1/sqrt(D) and log2(e) into Q: softmax runs in the exp2 domain.
    const float scale = 0.125f * 1.44269504088896340736f;

    _Float16* pw = ldsP + wave * 16 * P_STRIDE;     // this wave's P buffer

    // ---- Issue async global->LDS copies (fp32, B128) for one K/V tile.
    // 1024 B128 chunks per tile (K) + 1024 (V); 256 threads x 4 reps.
    // ASYNCcnt += 8 per wave per tile.
    auto issueTile = [&](int j0t, int bsel) {
        float* sk = stageK + bsel * TILE_F;
        float* sv = stageV + bsel * TILE_F;
#pragma unroll
        for (int rep = 0; rep < 4; ++rep) {
            const int c  = rep * 256 + t;          // B128 chunk id, 0..1023
            const int d  = c >> 4;                 // 16 chunks per 64-float row
            const int jf = (c & 15) * 4;           // first float within row
            const float* gk = &kp[base + (size_t)d * kN + j0t + jf];
            const float* gv = &vp[base + (size_t)d * kN + j0t + jf];
            const uint32_t lk = (uint32_t)(uintptr_t)&sk[d * 64 + jf];
            const uint32_t lv = (uint32_t)(uintptr_t)&sv[d * 64 + jf];
            asm volatile("global_load_async_to_lds_b128 %0, %1, off"
                         :: "v"(lk), "v"(gk) : "memory");
            asm volatile("global_load_async_to_lds_b128 %0, %1, off"
                         :: "v"(lv), "v"(gv) : "memory");
        }
    };

    // ---- Load Q A-fragments (two d-chunks of 32), scale folded in.
    // A 16x32 f16 layout: lane ln holds row i0+ln; vgpr<4 -> K in [0,15]
    // (+8 for upper lane half), vgpr>=4 -> K in [16,31] (+8 for upper half).
    AFrag qa[2];
#pragma unroll
    for (int c = 0; c < 2; ++c) {
#pragma unroll
        for (int e = 0; e < 16; ++e) {
            const int vr = e >> 1, p = e & 1;
            const int kk = (vr < 4) ? (lh * 8 + vr * 2 + p)
                                    : (16 + lh * 8 + (vr - 4) * 2 + p);
            const int d  = c * 32 + kk;
            qa[c].h[e] = (_Float16)(qp[base + (size_t)d * kN + (i0 + ln)] * scale);
        }
    }

    // Kick off tile 0 while Q loads are in flight.
    issueTile(0, 0);

    // ---- Online softmax state + output accumulators (C layout: row r / r+8).
    CFrag acc[4];
    float mrow[8], lrow[8];
#pragma unroll
    for (int dd = 0; dd < 4; ++dd) acc[dd].v = (v8f){};
#pragma unroll
    for (int r = 0; r < 8; ++r) { mrow[r] = -INFINITY; lrow[r] = 0.0f; }

    // =======================================================================
    // Stream over key/value tiles of 64 columns, software-pipelined.
    // =======================================================================
    int buf = 0;
    for (int j0 = 0; j0 < kN; j0 += kKT, buf ^= 1) {
        const bool hasNext = (j0 + kKT) < kN;
        if (hasNext) {
            issueTile(j0 + kKT, buf ^ 1);          // prefetch next tile async
            // 8 newer ops outstanding -> <=8 means current tile has landed
            asm volatile("s_wait_asynccnt 0x8" ::: "memory");
        } else {
            asm volatile("s_wait_asynccnt 0x0" ::: "memory");
        }
        __syncthreads();   // current tile visible to all waves; f16 bufs free

        const float* sk = stageK + buf * TILE_F;
        const float* sv = stageV + buf * TILE_F;

        // ---- Shared fp32 -> f16 conversion pass (packed cvt, b32 stores).
        // K tile also transposed: ldsKT[j][d] so B-frags load as (d,d+1) pairs.
#pragma unroll
        for (int rep = 0; rep < 8; ++rep) {
            const int u = rep * 256 + t;           // 0..2047
            const int dk = u >> 6, jk = u & 63;    // d-pair, j for K^T
            const float a0 = sk[(2 * dk)     * 64 + jk];
            const float a1 = sk[(2 * dk + 1) * 64 + jk];
            *(uint32_t*)&ldsKT[jk * KT_STRIDE + 2 * dk] =
                __builtin_bit_cast(uint32_t, __builtin_amdgcn_cvt_pkrtz(a0, a1));
            const int dv = u >> 5, jv = (u & 31) * 2;  // d, j-pair for V
            const float b0 = sv[dv * 64 + jv];
            const float b1 = sv[dv * 64 + jv + 1];
            *(uint32_t*)&ldsVh[dv * V_STRIDE + jv] =
                __builtin_bit_cast(uint32_t, __builtin_amdgcn_cvt_pkrtz(b0, b1));
        }
        __syncthreads();

        // ---- S = (Q*scale)^T K : four 16x16 j-subtiles, K-dim = 64.
        CFrag s[4];
#pragma unroll
        for (int jj = 0; jj < 4; ++jj) {
            AFrag bk[2];
#pragma unroll
            for (int c = 0; c < 2; ++c) {
#pragma unroll
                for (int vr = 0; vr < 8; ++vr) {
                    // B 32x16 f16: lane ln = column j; vgpr<4 -> K in [0,15]
                    // (+8 for upper lane half), vgpr>=4 -> K in [16,31] (+8 for upper half).
                    const int kk = (vr < 4) ? (lh * 8 + vr * 2)
                                            : (16 + lh * 8 + (vr - 4) * 2);
                    const int d  = c * 32 + kk;
                    const int j  = jj * 16 + ln;
                    bk[c].u[vr] = *(const uint32_t*)&ldsKT[j * KT_STRIDE + d];
                }
            }
            v8f z = (v8f){};
            z = __builtin_amdgcn_wmma_f32_16x16x32_f16(false, qa[0].v, false, bk[0].v,
                                                       (short)0, z, false, false);
            z = __builtin_amdgcn_wmma_f32_16x16x32_f16(false, qa[1].v, false, bk[1].v,
                                                       (short)0, z, false, false);
            s[jj].v = z;
        }

        // ---- Online softmax in exp2 domain. C layout: lane ln = column j,
        // vgpr r = row r (lanes 0-15) or r+8 (lanes 16-31). Row reductions
        // are xor-shuffles over the 16-lane group (masks 1..8 stay in-group).
        float rmax[8];
#pragma unroll
        for (int r = 0; r < 8; ++r) {
            float mx = fmaxf(fmaxf(s[0].f[r], s[1].f[r]),
                             fmaxf(s[2].f[r], s[3].f[r]));
#pragma unroll
            for (int m = 1; m < 16; m <<= 1)
                mx = fmaxf(mx, __shfl_xor(mx, m, 32));
            rmax[r] = mx;
        }
        float corr[8];
#pragma unroll
        for (int r = 0; r < 8; ++r) {
            const float mn = fmaxf(mrow[r], rmax[r]);
            corr[r] = __builtin_amdgcn_exp2f(mrow[r] - mn);  // exp2(-inf)=0
            mrow[r] = mn;
        }
        float rsum[8];
#pragma unroll
        for (int r = 0; r < 8; ++r) rsum[r] = 0.0f;
#pragma unroll
        for (int jj = 0; jj < 4; ++jj) {
#pragma unroll
            for (int r = 0; r < 8; ++r) {
                const float p = __builtin_amdgcn_exp2f(s[jj].f[r] - mrow[r]);
                rsum[r] += p;
                const int i = lh ? (r + 8) : r;
                pw[i * P_STRIDE + jj * 16 + ln] = (_Float16)p;
            }
        }
#pragma unroll
        for (int r = 0; r < 8; ++r) {
#pragma unroll
            for (int m = 1; m < 16; m <<= 1)
                rsum[r] += __shfl_xor(rsum[r], m, 32);
            lrow[r] = lrow[r] * corr[r] + rsum[r];
        }
#pragma unroll
        for (int dd = 0; dd < 4; ++dd)
#pragma unroll
            for (int r = 0; r < 8; ++r)
                acc[dd].f[r] *= corr[r];

        // ---- O += P V^T : reload P in A layout (per-wave LDS, DS in-order),
        // V fragments as B 32x16 (rows = j, cols = d).
        AFrag pa[2];
#pragma unroll
        for (int jb = 0; jb < 2; ++jb) {
#pragma unroll
            for (int vr = 0; vr < 8; ++vr) {
                const int koff = (vr < 4) ? (lh * 8 + vr * 2)
                                          : (16 + lh * 8 + (vr - 4) * 2);
                pa[jb].u[vr] =
                    *(const uint32_t*)&pw[ln * P_STRIDE + jb * 32 + koff];
            }
        }
#pragma unroll
        for (int dd = 0; dd < 4; ++dd) {
#pragma unroll
            for (int jb = 0; jb < 2; ++jb) {
                AFrag bv;
#pragma unroll
                for (int vr = 0; vr < 8; ++vr) {
                    const int kk = (vr < 4) ? (lh * 8 + vr * 2)
                                            : (16 + lh * 8 + (vr - 4) * 2);     // j within 32-chunk
                    bv.u[vr] = *(const uint32_t*)
                        &ldsVh[(dd * 16 + ln) * V_STRIDE + jb * 32 + kk];
                }
                acc[dd].v = __builtin_amdgcn_wmma_f32_16x16x32_f16(
                    false, pa[jb].v, false, bv.v, (short)0, acc[dd].v, false, false);
            }
        }
    }

    // =======================================================================
    // Epilogue: O[i][d]/l[i] -> out[d][i]. Transpose through LDS (stage
    // region, now idle) so global stores are contiguous 64B b128 along n.
    // =======================================================================
    __syncthreads();   // all async traffic drained; stage region reusable
    float* ow = ldsO + wave * kD * O_STRIDE;
#pragma unroll
    for (int dd = 0; dd < 4; ++dd) {
#pragma unroll
        for (int r = 0; r < 8; ++r) {
            const int i = lh ? (r + 8) : r;
            const int d = dd * 16 + ln;      // C layout: lane ln = column d
            ow[d * O_STRIDE + i] = acc[dd].f[r] / lrow[r];
        }
    }
    // Per-wave buffer + in-order DS: no barrier needed before readback.
#pragma unroll
    for (int rep = 0; rep < 2; ++rep) {
        const int d = rep * 32 + lane;       // each lane owns one d-row
        const float* src = &ow[d * O_STRIDE];
        float* dst = &outp[base + (size_t)d * kN + i0];
        const float4 x0 = *(const float4*)(src + 0);
        const float4 x1 = *(const float4*)(src + 4);
        const float4 x2 = *(const float4*)(src + 8);
        const float4 x3 = *(const float4*)(src + 12);
        *(float4*)(dst + 0)  = x0;
        *(float4*)(dst + 4)  = x1;
        *(float4*)(dst + 8)  = x2;
        *(float4*)(dst + 12) = x3;
    }
}

extern "C" void kernel_launch(void* const* d_in, const int* in_sizes, int n_in,
                              void* d_out, int out_size, void* d_ws, size_t ws_size,
                              hipStream_t stream)
{
    (void)in_sizes; (void)n_in; (void)out_size; (void)d_ws; (void)ws_size;
    const float* q = (const float*)d_in[0];
    const float* k = (const float*)d_in[1];
    const float* v = (const float*)d_in[2];
    float* out = (float*)d_out;

    dim3 block(256);
    dim3 grid(kB * kH * (kN / kIT));   // 64 heads x 16 query tiles = 1024 blocks
    fa_fwd_kernel<<<grid, block, 0, stream>>>(q, k, v, out);
}
